// NeuralBellmanFordNetwork_75737453298301
// MI455X (gfx1250) — matrix operand order repaired and checked by reading, compiled, not run on hardware
//
#include <hip/hip_runtime.h>
#include <math.h>

// ---------------- problem constants (from reference) ----------------
#define N_NODES 25000
#define N_EDGES 500000
#define BQ 4
#define DIM 32
#define R2C 474
#define LAY 4
#define K13 (13 * DIM)          // 416
#define NBD (N_NODES * BQ * DIM) // 3,200,000
#define M_ROWS (N_NODES * BQ)    // 100,000  (divisible by 16)
#define EPSV 1e-6f

typedef __attribute__((ext_vector_type(16))) _Float16 v16h;
typedef __attribute__((ext_vector_type(8)))  float    v8f;

// ---------------- small helpers ----------------
__device__ __forceinline__ void atomicMaxF(float* a, float v) {
  if (v >= 0.0f) atomicMax((int*)a, __float_as_int(v));
  else           atomicMin((unsigned int*)a, __float_as_uint(v));
}
__device__ __forceinline__ void atomicMinF(float* a, float v) {
  if (v >= 0.0f) atomicMin((int*)a, __float_as_int(v));
  else           atomicMax((unsigned int*)a, __float_as_uint(v));
}

__global__ void fillf(float* p, float v, int n) {
  int i = blockIdx.x * 256 + threadIdx.x;
  if (i < n) p[i] = v;
}

__global__ void cvt_half(const float* src, _Float16* dst, int n) {
  int i = blockIdx.x * 256 + threadIdx.x;
  if (i < n) dst[i] = (_Float16)src[i];
}

// query gather + boundary scatter.  <<<1, BQ*DIM>>>
__global__ void scatter_boundary(const int* __restrict__ h_index,
                                 const int* __restrict__ r_index,
                                 const float* __restrict__ query_embed,
                                 float* __restrict__ query,
                                 float* __restrict__ boundary) {
  int i = threadIdx.x;          // 0..127
  int b = i >> 5, d = i & 31;
  float q = query_embed[r_index[b] * DIM + d];
  query[b * DIM + d] = q;
  boundary[(h_index[b] * BQ + b) * DIM + d] = q;
}

// per-node incoming count + weighted degree
__global__ void edge_count(const int* __restrict__ node_out,
                           const float* __restrict__ ew,
                           float* __restrict__ cnt, float* __restrict__ deg) {
  int e = blockIdx.x * 256 + threadIdx.x;
  if (e < N_EDGES) {
    int no = node_out[e];
    atomicAdd(cnt + no, 1.0f);
    atomicAdd(deg + no, ew[e]);
  }
}

// scale = log(deg+1); accumulate global sum; cnt += 1 (self message)
__global__ void scale_pass1(const float* __restrict__ deg, float* __restrict__ cnt,
                            float* __restrict__ slog, float* __restrict__ accum) {
  __shared__ float red[256];
  int n = blockIdx.x * 256 + threadIdx.x;
  float s = 0.0f;
  if (n < N_NODES) {
    float dd = deg[n] + 1.0f;
    cnt[n] += 1.0f;
    s = logf(dd);
    slog[n] = s;
  }
  red[threadIdx.x] = s;
  __syncthreads();
  for (int off = 128; off > 0; off >>= 1) {
    if (threadIdx.x < off) red[threadIdx.x] += red[threadIdx.x + off];
    __syncthreads();
  }
  if (threadIdx.x == 0) atomicAdd(accum, red[0]);
}

__global__ void scale_pass2(const float* __restrict__ slog,
                            const float* __restrict__ accum,
                            float* __restrict__ scales) {
  int n = blockIdx.x * 256 + threadIdx.x;
  if (n >= N_NODES) return;
  float mean = accum[0] / (float)N_NODES;
  float s = slog[n] / mean;
  scales[n * 3 + 0] = 1.0f;
  scales[n * 3 + 1] = s;
  scales[n * 3 + 2] = 1.0f / fmaxf(s, 0.01f);
}

// rel_in[r,b,d] = sum_k query[b,k] * rel_W[l][k, r*D+d] + rel_b[l][r*D+d]
__global__ void rel_in_kernel(const float* __restrict__ query,
                              const float* __restrict__ relW,  // layer slice [DIM][R2C*DIM]
                              const float* __restrict__ relb,  // layer slice [R2C*DIM]
                              float* __restrict__ rel_in) {    // [R2C][BQ][DIM]
  int col = blockIdx.x * 256 + threadIdx.x;
  if (col >= R2C * DIM) return;
  float acc0 = relb[col], acc1 = acc0, acc2 = acc0, acc3 = acc0;
  #pragma unroll 8
  for (int k = 0; k < DIM; ++k) {
    float w = relW[(size_t)k * (R2C * DIM) + col];
    acc0 += query[0 * DIM + k] * w;
    acc1 += query[1 * DIM + k] * w;
    acc2 += query[2 * DIM + k] * w;
    acc3 += query[3 * DIM + k] * w;
  }
  int r = col / DIM, d = col - r * DIM;
  rel_in[((r * BQ + 0)) * DIM + d] = acc0;
  rel_in[((r * BQ + 1)) * DIM + d] = acc1;
  rel_in[((r * BQ + 2)) * DIM + d] = acc2;
  rel_in[((r * BQ + 3)) * DIM + d] = acc3;
}

// accumulators start with the self/boundary message (weight = 1)
__global__ void init_accum(const float* __restrict__ boundary,
                           float* __restrict__ sum, float* __restrict__ sq,
                           float* __restrict__ mx, float* __restrict__ mn) {
  int i = blockIdx.x * 256 + threadIdx.x;
  if (i < NBD) {
    float v = boundary[i];
    sum[i] = v; sq[i] = v * v; mx[i] = v; mn[i] = v;
  }
}

// DistMult message + 4 segment reductions. 2 edges per 256-thread block.
__global__ void msg_kernel(const int* __restrict__ node_in, const int* __restrict__ node_out,
                           const int* __restrict__ relation, const float* __restrict__ ew,
                           const float* __restrict__ rel_in, const float* __restrict__ hidden,
                           float* __restrict__ sum, float* __restrict__ sq,
                           float* __restrict__ mx, float* __restrict__ mn) {
  int e = blockIdx.x * 2 + (threadIdx.x >> 7);
  if (e >= N_EDGES) return;
  int lane = threadIdx.x & 127;
  int b = lane >> 5, d = lane & 31;
  int ni = node_in[e], no = node_out[e], r = relation[e];
  float w = ew[e];
  float m  = rel_in[(r * BQ + b) * DIM + d] * hidden[(ni * BQ + b) * DIM + d];
  float mw = m * w;
  int base = (no * BQ + b) * DIM + d;
  atomicAdd(sum + base, mw);
  atomicAdd(sq  + base, m * mw);
  atomicMaxF(mx + base, mw);
  atomicMinF(mn + base, mw);
}

// ------------- per-layer fused PNA-feature + [16x416]x[416x32] WMMA GEMM -------------
// one wave32 per 16-row M tile; 13 k-steps x 2 n-tiles of v_wmma_f32_16x16x32_f16
__global__ __launch_bounds__(32) void layer_gemm(
    const float* __restrict__ hidden, const float* __restrict__ sum,
    const float* __restrict__ sq, const float* __restrict__ mx,
    const float* __restrict__ mn, const float* __restrict__ cnt,
    const float* __restrict__ scales, const _Float16* __restrict__ wh, // [K13][DIM] f16
    const float* __restrict__ bias, float* __restrict__ hnext) {
  __shared__ _Float16 xs[16][K13 + 8];
  int t = threadIdx.x;
  int row0 = blockIdx.x * 16;

  // stage the 16x416 A tile (hidden | PNA-update) into LDS as f16
  for (int i = t; i < 16 * K13; i += 32) {
    int r = i / K13, c = i - r * K13;
    int row = row0 + r;              // row = n*BQ + b
    int n = row >> 2;
    float v;
    if (c < DIM) {
      v = hidden[row * DIM + c];
    } else {
      int u = c - DIM;
      int d = u / 12;
      int rem = u - d * 12;
      int s = rem / 3;
      int ksc = rem - s * 3;
      int base = row * DIM + d;
      float cn = cnt[n];
      float stat;
      if (s == 0) stat = sum[base] / cn;
      else if (s == 1) stat = mx[base];
      else if (s == 2) stat = mn[base];
      else {
        float me = sum[base] / cn;
        stat = sqrtf(fmaxf(sq[base] / cn - me * me, EPSV));
      }
      v = stat * scales[n * 3 + ksc];
    }
    xs[r][c] = (_Float16)v;
  }
  __syncthreads();

  v8f c0 = {}; v8f c1 = {};
  int m = t & 15;
  int hi = (t >> 4) * 8;
  for (int ks = 0; ks < 13; ++ks) {
    int kb = ks * 32;
    v16h a;
    #pragma unroll
    for (int j = 0; j < 8; ++j) {
      a[j]     = xs[m][kb + hi + j];          // K = hi + j
      a[j + 8] = xs[m][kb + 16 + hi + j];     // K = 16 + hi + j
    }
    v16h b0, b1;
    #pragma unroll
    for (int j = 0; j < 8; ++j) {
      b0[j]     = wh[(size_t)(kb + hi + j) * DIM + m];
      b0[j + 8] = wh[(size_t)(kb + 16 + hi + j) * DIM + m];
      b1[j]     = wh[(size_t)(kb + hi + j) * DIM + 16 + m];
      b1[j + 8] = wh[(size_t)(kb + 16 + hi + j) * DIM + 16 + m];
    }
    c0 = __builtin_amdgcn_wmma_f32_16x16x32_f16(false, a, false, b0, (short)0, c0, false, false);
    c1 = __builtin_amdgcn_wmma_f32_16x16x32_f16(false, a, false, b1, (short)0, c1, false, false);
  }

  // epilogue: bias + relu + store  (C layout: M = r + (t>>4)*8, N = t&15)
  int nc = t & 15;
  int mb = (t >> 4) * 8;
  float bv0 = bias[nc];
  float bv1 = bias[16 + nc];
  #pragma unroll
  for (int r = 0; r < 8; ++r) {
    int row = row0 + mb + r;
    hnext[row * DIM + nc]      = fmaxf(c0[r] + bv0, 0.0f);
    hnext[row * DIM + 16 + nc] = fmaxf(c1[r] + bv1, 0.0f);
  }
}

// ------------- final MLP: [16x64] @ [64x64] WMMA, relu, dot with W2 -------------
__global__ __launch_bounds__(32) void final_kernel(
    const float* __restrict__ hidden, const float* __restrict__ query,
    const _Float16* __restrict__ w1h, const float* __restrict__ b1,
    const float* __restrict__ w2, const float* __restrict__ b2,
    float* __restrict__ out) {
  __shared__ _Float16 xs[16][72];
  __shared__ float h1s[16][68];
  int t = threadIdx.x;
  int row0 = blockIdx.x * 16;

  for (int i = t; i < 16 * 64; i += 32) {
    int r = i >> 6, c = i & 63;
    int row = row0 + r;
    int b = row & 3;
    float v = (c < DIM) ? hidden[row * DIM + c] : query[b * DIM + (c - DIM)];
    xs[r][c] = (_Float16)v;
  }
  __syncthreads();

  v8f a0 = {}; v8f a1 = {}; v8f a2 = {}; v8f a3 = {};
  int m = t & 15;
  int hi = (t >> 4) * 8;
  #pragma unroll
  for (int ks = 0; ks < 2; ++ks) {
    int kb = ks * 32;
    v16h a;
    #pragma unroll
    for (int j = 0; j < 8; ++j) {
      a[j]     = xs[m][kb + hi + j];
      a[j + 8] = xs[m][kb + 16 + hi + j];
    }
    v16h bp[4];
    #pragma unroll
    for (int q = 0; q < 4; ++q) {
      #pragma unroll
      for (int j = 0; j < 8; ++j) {
        bp[q][j]     = w1h[(size_t)(kb + hi + j) * 64 + q * 16 + m];
        bp[q][j + 8] = w1h[(size_t)(kb + 16 + hi + j) * 64 + q * 16 + m];
      }
    }
    a0 = __builtin_amdgcn_wmma_f32_16x16x32_f16(false, a, false, bp[0], (short)0, a0, false, false);
    a1 = __builtin_amdgcn_wmma_f32_16x16x32_f16(false, a, false, bp[1], (short)0, a1, false, false);
    a2 = __builtin_amdgcn_wmma_f32_16x16x32_f16(false, a, false, bp[2], (short)0, a2, false, false);
    a3 = __builtin_amdgcn_wmma_f32_16x16x32_f16(false, a, false, bp[3], (short)0, a3, false, false);
  }

  int nc = t & 15;
  int mb = (t >> 4) * 8;
  #pragma unroll
  for (int r = 0; r < 8; ++r) {
    h1s[mb + r][nc]      = fmaxf(a0[r] + b1[nc],      0.0f);
    h1s[mb + r][16 + nc] = fmaxf(a1[r] + b1[16 + nc], 0.0f);
    h1s[mb + r][32 + nc] = fmaxf(a2[r] + b1[32 + nc], 0.0f);
    h1s[mb + r][48 + nc] = fmaxf(a3[r] + b1[48 + nc], 0.0f);
  }
  __syncthreads();

  if (t < 16) {
    float acc = b2[0];
    #pragma unroll 8
    for (int j = 0; j < 64; ++j) acc += h1s[t][j] * w2[j];
    int row = row0 + t;
    int n = row >> 2, b = row & 3;
    out[b * N_NODES + n] = acc;
  }
}

// ---------------- host ----------------
extern "C" void kernel_launch(void* const* d_in, const int* in_sizes, int n_in,
                              void* d_out, int out_size, void* d_ws, size_t ws_size,
                              hipStream_t stream) {
  (void)in_sizes; (void)n_in; (void)out_size; (void)ws_size;
  const int*   node_in     = (const int*)d_in[0];
  const int*   node_out    = (const int*)d_in[1];
  const int*   relation    = (const int*)d_in[2];
  const float* edge_weight = (const float*)d_in[3];
  const int*   h_index     = (const int*)d_in[4];
  const int*   r_index     = (const int*)d_in[5];
  const float* query_embed = (const float*)d_in[6];
  const float* rel_W       = (const float*)d_in[7];
  const float* rel_b       = (const float*)d_in[8];
  const float* lin_W       = (const float*)d_in[9];
  const float* lin_b       = (const float*)d_in[10];
  const float* mlp_W1      = (const float*)d_in[11];
  const float* mlp_b1      = (const float*)d_in[12];
  const float* mlp_W2      = (const float*)d_in[13];
  const float* mlp_b2      = (const float*)d_in[14];
  float* out = (float*)d_out;

  // workspace carve-up (256B aligned)
  size_t off = 0;
  auto alloc = [&](size_t bytes) -> char* {
    char* p = (char*)d_ws + off;
    off = (off + bytes + 255) & ~(size_t)255;
    return p;
  };
  float* boundary = (float*)alloc(sizeof(float) * NBD);
  float* h0       = (float*)alloc(sizeof(float) * NBD);
  float* h1       = (float*)alloc(sizeof(float) * NBD);
  float* sum      = (float*)alloc(sizeof(float) * NBD);
  float* sq       = (float*)alloc(sizeof(float) * NBD);
  float* mx       = (float*)alloc(sizeof(float) * NBD);
  float* mn       = (float*)alloc(sizeof(float) * NBD);
  float* cnt      = (float*)alloc(sizeof(float) * N_NODES);
  float* deg      = (float*)alloc(sizeof(float) * N_NODES);
  float* slog     = (float*)alloc(sizeof(float) * N_NODES);
  float* scales   = (float*)alloc(sizeof(float) * N_NODES * 3);
  float* query    = (float*)alloc(sizeof(float) * BQ * DIM);
  float* rel_in   = (float*)alloc(sizeof(float) * R2C * BQ * DIM);
  float* accum    = (float*)alloc(sizeof(float) * 4);
  _Float16* linWh = (_Float16*)alloc(sizeof(_Float16) * LAY * K13 * DIM);
  _Float16* w1h   = (_Float16*)alloc(sizeof(_Float16) * 64 * 64);

  auto G = [](int n, int b) { return (n + b - 1) / b; };

  // setup
  fillf<<<G(NBD, 256), 256, 0, stream>>>(boundary, 0.0f, NBD);
  fillf<<<G(N_NODES, 256), 256, 0, stream>>>(cnt, 0.0f, N_NODES);
  fillf<<<G(N_NODES, 256), 256, 0, stream>>>(deg, 0.0f, N_NODES);
  fillf<<<1, 256, 0, stream>>>(accum, 0.0f, 1);
  scatter_boundary<<<1, BQ * DIM, 0, stream>>>(h_index, r_index, query_embed, query, boundary);
  edge_count<<<G(N_EDGES, 256), 256, 0, stream>>>(node_out, edge_weight, cnt, deg);
  scale_pass1<<<G(N_NODES, 256), 256, 0, stream>>>(deg, cnt, slog, accum);
  scale_pass2<<<G(N_NODES, 256), 256, 0, stream>>>(slog, accum, scales);
  cvt_half<<<G(LAY * K13 * DIM, 256), 256, 0, stream>>>(lin_W, linWh, LAY * K13 * DIM);
  cvt_half<<<G(64 * 64, 256), 256, 0, stream>>>(mlp_W1, w1h, 64 * 64);

  // layers
  const float* hcur = boundary;
  float* hbuf[2] = {h0, h1};
  for (int l = 0; l < LAY; ++l) {
    rel_in_kernel<<<G(R2C * DIM, 256), 256, 0, stream>>>(
        query, rel_W + (size_t)l * DIM * R2C * DIM, rel_b + (size_t)l * R2C * DIM, rel_in);
    init_accum<<<G(NBD, 256), 256, 0, stream>>>(boundary, sum, sq, mx, mn);
    msg_kernel<<<N_EDGES / 2, 256, 0, stream>>>(node_in, node_out, relation, edge_weight,
                                                rel_in, hcur, sum, sq, mx, mn);
    layer_gemm<<<M_ROWS / 16, 32, 0, stream>>>(hcur, sum, sq, mx, mn, cnt, scales,
                                               linWh + (size_t)l * K13 * DIM,
                                               lin_b + (size_t)l * DIM, hbuf[l & 1]);
    hcur = hbuf[l & 1];
  }

  // final MLP + scores
  final_kernel<<<M_ROWS / 16, 32, 0, stream>>>(hcur, query, w1h, mlp_b1, mlp_W2, mlp_b2, out);
}
